// PiT_28913719837190
// MI455X (gfx1250) — matrix operand order repaired and checked by reading, not compiled, not run on hardware
//
#include <hip/hip_runtime.h>
#include <hip/hip_fp16.h>
#include <cmath>

typedef __attribute__((ext_vector_type(16))) _Float16 v16h;
typedef __attribute__((ext_vector_type(8)))  float    v8f;

enum {
  EPI_BIAS = 1,
  EPI_GELU = 2,
  EPI_ADD  = 4,
  EPI_F32  = 8,
  EPI_F16  = 16,
  SC_VALUE = 32,
  SC_ATTN  = 64,
};

__device__ __forceinline__ float gelu_f(float x) {
  return 0.5f * x * (1.0f + erff(x * 0.70710678118654752f));
}

union FragH { v16h v; unsigned int u[8]; };

#define TM 128
#define TN 128
#define KC 32

// ---- CDNA5 async global->LDS copy (ASYNCcnt path), with safe fallback ----
typedef int ai4 __attribute__((ext_vector_type(4)));
typedef ai4 __attribute__((address_space(1)))* gp4;  // global int4*
typedef ai4 __attribute__((address_space(3)))* lp4;  // LDS int4*

#if __has_builtin(__builtin_amdgcn_global_load_async_to_lds_b128)
#define HAVE_ASYNC_LDS 1
// Declare the LLVM intrinsic directly so we control the parameter types
// (the clang builtin wants __device__/__shared__ language-AS pointers that
// cannot be spelled in HIP source). Opaque pointers: only AS1/AS3 must match.
extern "C" __device__ void
__cdna5_async_load_b128(gp4 gsrc, lp4 ldst, int off, int cpol)
    __asm("llvm.amdgcn.global.load.async.to.lds.b128");
#endif

__device__ __forceinline__ void cp_async_b128(__half* lds, const __half* g) {
#ifdef HAVE_ASYNC_LDS
  // inttoptr casts: flat global low 64b are the AS1 address; the flat-shared
  // aperture keeps the LDS byte offset in the low 32 bits for the AS3 pointer.
  __cdna5_async_load_b128((gp4)(__UINTPTR_TYPE__)(g),
                          (lp4)(__UINTPTR_TYPE__)(lds), 0, 0);
#else
  *(uint4*)lds = *(const uint4*)g;
#endif
}

#if __has_builtin(__builtin_amdgcn_s_wait_asynccnt)
#define WAIT_ASYNC(n) __builtin_amdgcn_s_wait_asynccnt(n)
#else
#define WAIT_ASYNC(n) asm volatile("s_wait_asynccnt %0" ::"i"(n) : "memory")
#endif

// Generic f16 WMMA GEMM: C[M, Ncols] = A[M,K] * B[K,Ncols], B given transposed
// (BT[col][k], leading dim ldb). 256 threads = 8 waves; each wave owns a
// 16-row strip and 8 column tiles (128 cols per workgroup, blockIdx.y picks
// which 128). K staged through double-buffered LDS in chunks of 32 (one WMMA
// K-step) using async global->LDS DMA so the copy overlaps the WMMA stream.
__global__ __launch_bounds__(256) void k_wmma_gemm(
    const __half* __restrict__ A, int lda,
    const __half* __restrict__ BT, int ldb,
    int M, int K,
    const float* __restrict__ bias,
    const float* __restrict__ other,
    float* __restrict__ outF,
    __half* __restrict__ outH,
    int ldo, int mode, int head)
{
  __shared__ __half sA[2][TM * KC];
  __shared__ __half sB[2][TN * KC];
  const int tid  = threadIdx.x;
  const int wave = tid >> 5;
  const int lane = tid & 31;
  const int hv   = lane >> 4;   // lane half (0: lanes 0-15, 1: lanes 16-31)
  const int r    = lane & 15;   // row (A) / col (B/C/D) within 16
  const int rowBase = blockIdx.x * TM;
  const int colBase = blockIdx.y * TN;

  v8f acc[8];
  #pragma unroll
  for (int i = 0; i < 8; ++i)
    #pragma unroll
    for (int e = 0; e < 8; ++e) acc[i][e] = 0.0f;

  // Per stage: each wave issues 2 A-chunk + 2 B-chunk async b128 loads.
  auto stage = [&](int buf, int k0) {
    for (int ci = tid; ci < TM * KC / 8; ci += 256) {
      int row = ci >> 2, seg = ci & 3;
      cp_async_b128(sA[buf] + row * KC + seg * 8,
                    A + (size_t)(rowBase + row) * lda + k0 + seg * 8);
    }
    for (int ci = tid; ci < TN * KC / 8; ci += 256) {
      int col = ci >> 2, seg = ci & 3;
      cp_async_b128(sB[buf] + col * KC + seg * 8,
                    BT + (size_t)(colBase + col) * ldb + k0 + seg * 8);
    }
  };

  const int nsteps = K / KC;
  stage(0, 0);

  for (int it = 0; it < nsteps; ++it) {
    const int cur = it & 1;
    __syncthreads();  // all waves done reading buf[cur^1] from previous step
    if (it + 1 < nsteps) {
      stage(cur ^ 1, (it + 1) * KC);
      WAIT_ASYNC(4);  // 4 per-wave loads of the NEXT tile may stay in flight
    } else {
      WAIT_ASYNC(0);
    }
    __syncthreads();  // every wave's DMA for buf[cur] has landed

    // A fragment: row = r; lanes 0-15 hold K 0-7 & 16-23, lanes 16-31 hold
    // K 8-15 & 24-31 (ISA 7.12.2, 16-bit A 16x32).
    FragH fa;
    {
      const unsigned int* base =
          (const unsigned int*)(sA[cur] + (wave * 16 + r) * KC + hv * 8);
      #pragma unroll
      for (int p = 0; p < 4; ++p) { fa.u[p] = base[p]; fa.u[p + 4] = base[p + 8]; }
    }
    #pragma unroll
    for (int nt = 0; nt < 8; ++nt) {
      // B fragment: col = r; lanes 0-15 hold K 0-7 & 16-23, lanes 16-31 hold K 8-15 & 24-31.
      FragH fb;
      const unsigned int* base =
          (const unsigned int*)(sB[cur] + (nt * 16 + r) * KC + hv * 8);
      #pragma unroll
      for (int p = 0; p < 4; ++p) { fb.u[p] = base[p]; fb.u[p + 4] = base[p + 8]; }
      acc[nt] = __builtin_amdgcn_wmma_f32_16x16x32_f16(
          false, fa.v, false, fb.v, (short)0, acc[nt], false, false);
    }
  }

  // D layout: VGPR v -> M = v + hv*8, N = r (ISA 7.12.2, 32-bit C/D 16x16).
  #pragma unroll
  for (int nt = 0; nt < 8; ++nt) {
    int cg = colBase + nt * 16 + r;
    #pragma unroll
    for (int v = 0; v < 8; ++v) {
      int rg = rowBase + wave * 16 + v + hv * 8;
      if (rg >= M) continue;
      float val = acc[nt][v];
      if (mode & EPI_BIAS) val += bias[cg];
      if (mode & EPI_ADD)  val += other[(size_t)rg * ldo + cg];
      if (mode & EPI_GELU) val = gelu_f(val);
      if (mode & SC_VALUE) {
        // row = b*2048+n, col = h*32+k  ->  valT[h*128 + b*32 + k][n]
        int b  = rg >> 11, n  = rg & 2047;
        int h2 = cg >> 5,  k2 = cg & 31;
        outH[((size_t)(h2 * 128 + b * 32 + k2)) * 2048 + n] = __float2half(val);
      } else if (mode & SC_ATTN) {
        // row = j, col = b*32+k  ->  a[(b*2048+j)*256 + head*32 + k]
        int b = cg >> 5, k2 = cg & 31;
        outH[((size_t)(b * 2048 + rg)) * 256 + head * 32 + k2] = __float2half(val);
      } else {
        if (mode & EPI_F32) outF[(size_t)rg * ldo + cg] = val;
        if (mode & EPI_F16) outH[(size_t)rg * ldo + cg] = __float2half(val);
      }
    }
  }
}

// Encoder: h = gelu(x @ en_w + en_b), K = 3, output f16 [8192,256].
__global__ void k_encoder(const float* __restrict__ x,
                          const float* __restrict__ w,
                          const float* __restrict__ b,
                          __half* __restrict__ out) {
  int id = blockIdx.x * 256 + threadIdx.x;  // 8192*256 threads
  int bn = id >> 8, c = id & 255;
  float v = b[c];
  #pragma unroll
  for (int j = 0; j < 3; ++j) v += x[bn * 3 + j] * w[j * 256 + c];
  out[id] = __float2half(gelu_f(v));
}

// Weight f32 [256][256] -> f16 transposed WT[c][j].
__global__ void k_cvt_wT(const float* __restrict__ in, __half* __restrict__ out) {
  int c = blockIdx.x, j = threadIdx.x;
  out[c * 256 + j] = __float2half(in[j * 256 + c]);
}

// Head weights f32 [8][256][32] -> f16 WT[c = h*32+k][j] = w[h][j][k].
__global__ void k_cvt_wvT(const float* __restrict__ in, __half* __restrict__ out) {
  int c = blockIdx.x, j = threadIdx.x;
  int h = c >> 5, k = c & 31;
  out[c * 256 + j] = __float2half(in[(h * 256 + j) * 32 + k]);
}

// Per-row quantile threshold: bitonic sort of 2048 f32 in LDS, linear interp.
// (scale = tan(...) >= 0, so quantile of m*scale masks identically to
//  quantile of m; threshold is head-independent.)
__global__ __launch_bounds__(256) void k_thr(const float* __restrict__ m,
                                             float* __restrict__ thr, float q) {
  __shared__ float s[2048];
  int j = blockIdx.x, t = threadIdx.x;
  for (int i = t; i < 2048; i += 256) s[i] = m[(size_t)j * 2048 + i];
  __syncthreads();
  for (int k = 2; k <= 2048; k <<= 1) {
    for (int jj = k >> 1; jj > 0; jj >>= 1) {
      for (int i = t; i < 2048; i += 256) {
        int l = i ^ jj;
        if (l > i) {
          float a = s[i], b = s[l];
          bool up = ((i & k) == 0);
          if ((a > b) == up) { s[i] = b; s[l] = a; }
        }
      }
      __syncthreads();
    }
  }
  if (t == 0) {
    float idx = q * 2047.0f;
    int lo = (int)idx;
    int hi = lo + 1; if (hi > 2047) hi = 2047;
    float fr = idx - (float)lo;
    thr[j] = s[lo] + fr * (s[hi] - s[lo]);
  }
}

// One row of one head's attention: masked softmax of exp(-m*scale) -> f16.
__global__ __launch_bounds__(256) void k_att_row(const float* __restrict__ m,
                                                 const float* __restrict__ thr,
                                                 const float* __restrict__ rv,
                                                 int h, __half* __restrict__ att) {
  __shared__ float sm[2048];
  __shared__ float red[256];
  int j = blockIdx.x, t = threadIdx.x;
  float scale = tanf(0.78539816339744831f * (1.0f - 1e-7f) * (1.0f + sinf(rv[h])));
  float tj = thr[j];
  float sum = 0.0f;
  for (int i = t; i < 2048; i += 256) {
    float mv = m[(size_t)j * 2048 + i];
    sm[i] = mv;
    if (mv <= tj) sum += expf(-mv * scale);
  }
  red[t] = sum;
  __syncthreads();
  for (int s = 128; s > 0; s >>= 1) {
    if (t < s) red[t] += red[t + s];
    __syncthreads();
  }
  float inv = 1.0f / red[0];
  for (int i = t; i < 2048; i += 256) {
    float mv = sm[i];
    float a = (mv <= tj) ? expf(-mv * scale) * inv : 0.0f;
    att[(size_t)j * 2048 + i] = __float2half(a);
  }
}

// Final projection: out[row] = h[row,:] . de2_w + de2_b  (OUT_C = 1).
__global__ void k_de2(const __half* __restrict__ a, const float* __restrict__ w,
                      const float* __restrict__ b, float* __restrict__ out) {
  int row = blockIdx.x * 256 + threadIdx.x;
  float v = b[0];
  for (int c = 0; c < 256; ++c)
    v += __half2float(a[(size_t)row * 256 + c]) * w[c];
  out[row] = v;
}

extern "C" void kernel_launch(void* const* d_in, const int* in_sizes, int n_in,
                              void* d_out, int out_size, void* d_ws, size_t ws_size,
                              hipStream_t stream) {
  (void)in_sizes; (void)n_in; (void)out_size; (void)ws_size;
  const float* x       = (const float*)d_in[0];
  const float* m0      = (const float*)d_in[1];
  const float* m1      = (const float*)d_in[2];
  const float* m2      = (const float*)d_in[3];
  const float* m3      = (const float*)d_in[4];
  const float* en_w    = (const float*)d_in[5];
  const float* en_b    = (const float*)d_in[6];
  const float* down_r  = (const float*)d_in[7];
  const float* down_w  = (const float*)d_in[8];
  const float* pa0_r   = (const float*)d_in[9];
  const float* pa0_w   = (const float*)d_in[10];
  const float* pa1_r   = (const float*)d_in[11];
  const float* pa1_w   = (const float*)d_in[12];
  const float* mlp0_w1 = (const float*)d_in[13];
  const float* mlp0_b1 = (const float*)d_in[14];
  const float* mlp0_w2 = (const float*)d_in[15];
  const float* mlp0_b2 = (const float*)d_in[16];
  const float* w0_w    = (const float*)d_in[17];
  const float* w0_b    = (const float*)d_in[18];
  const float* mlp1_w1 = (const float*)d_in[19];
  const float* mlp1_b1 = (const float*)d_in[20];
  const float* mlp1_w2 = (const float*)d_in[21];
  const float* mlp1_b2 = (const float*)d_in[22];
  const float* w1_w    = (const float*)d_in[23];
  const float* w1_b    = (const float*)d_in[24];
  const float* up_r    = (const float*)d_in[25];
  const float* up_w    = (const float*)d_in[26];
  const float* de1_w   = (const float*)d_in[27];
  const float* de1_b   = (const float*)d_in[28];
  const float* de2_w   = (const float*)d_in[29];
  const float* de2_b   = (const float*)d_in[30];

  char* ws = (char*)d_ws;
  size_t off = 0;
  auto carve = [&](size_t bytes) -> char* {
    char* p = ws + off;
    off += (bytes + 255) & ~(size_t)255;
    return p;
  };
  __half* act0 = (__half*)carve(8192ull * 256 * 2);   // activation ping (f16)
  __half* act1 = (__half*)carve(8192ull * 256 * 2);   // activation pong (f16)
  __half* act2 = (__half*)carve(8192ull * 256 * 2);   // activation pang (f16)
  float*  g0   = (float*) carve(8192ull * 256 * 4);   // f32 residual-path buffer
  __half* valT = (__half*)carve(1024ull * 2048 * 2);  // valT[h*128+b*32+k][n]
  __half* att  = (__half*)carve(2048ull * 2048 * 2);  // one head's att, reused
  float*  thr  = (float*) carve(2048ull * 4);         // per-row quantile thr
  __half* wT[11];
  for (int i = 0; i < 11; ++i) wT[i] = (__half*)carve(256ull * 256 * 2);

  // Convert all weights to f16, transposed for the GEMM's BT layout.
  k_cvt_wvT<<<256, 256, 0, stream>>>(down_w,  wT[0]);
  k_cvt_wvT<<<256, 256, 0, stream>>>(pa0_w,   wT[1]);
  k_cvt_wvT<<<256, 256, 0, stream>>>(pa1_w,   wT[2]);
  k_cvt_wvT<<<256, 256, 0, stream>>>(up_w,    wT[3]);
  k_cvt_wT <<<256, 256, 0, stream>>>(mlp0_w1, wT[4]);
  k_cvt_wT <<<256, 256, 0, stream>>>(mlp0_w2, wT[5]);
  k_cvt_wT <<<256, 256, 0, stream>>>(w0_w,    wT[6]);
  k_cvt_wT <<<256, 256, 0, stream>>>(mlp1_w1, wT[7]);
  k_cvt_wT <<<256, 256, 0, stream>>>(mlp1_w2, wT[8]);
  k_cvt_wT <<<256, 256, 0, stream>>>(w1_w,    wT[9]);
  k_cvt_wT <<<256, 256, 0, stream>>>(de1_w,   wT[10]);

  k_encoder<<<8192, 256, 0, stream>>>(x, en_w, en_b, act0);

  auto gemm256 = [&](const __half* Ain, const __half* BTw, const float* bias,
                     const float* other, float* oF, __half* oH, int mode) {
    k_wmma_gemm<<<dim3(64, 2), 256, 0, stream>>>(Ain, 256, BTw, 256, 8192, 256,
                                                 bias, other, oF, oH, 256, mode, 0);
  };

  auto pos_att = [&](const float* md, const float* rv, const __half* WvT,
                     const __half* aIn, __half* aOut, float q) {
    k_thr<<<2048, 256, 0, stream>>>(md, thr, q);
    // value = act @ Wv  -> valT (f16, transposed per head for the next GEMM)
    k_wmma_gemm<<<dim3(64, 2), 256, 0, stream>>>(aIn, 256, WvT, 256, 8192, 256,
        nullptr, nullptr, nullptr, valT, 256, SC_VALUE, 0);
    for (int h = 0; h < 8; ++h) {
      k_att_row<<<2048, 256, 0, stream>>>(md, thr, rv, h, att);
      // out_h = gelu(att_h [2048,2048] @ V_h [2048,128]) scattered into aOut
      k_wmma_gemm<<<dim3(16, 1), 256, 0, stream>>>(att, 2048,
          valT + (size_t)h * 128 * 2048, 2048, 2048, 2048,
          nullptr, nullptr, nullptr, aOut, 256, SC_ATTN | EPI_GELU, h);
    }
  };

  // down pos_att
  pos_att(m0, down_r, wT[0], act0, act1, 0.50f);

  // block 0
  pos_att(m1, pa0_r, wT[1], act1, act2, 0.50f);
  gemm256(act2, wT[4], mlp0_b1, nullptr, nullptr, act0, EPI_BIAS | EPI_GELU | EPI_F16);
  gemm256(act1, wT[6], w0_b,    nullptr, g0,      nullptr, EPI_BIAS | EPI_F32);
  gemm256(act0, wT[5], mlp0_b2, g0,      nullptr, act2, EPI_BIAS | EPI_ADD | EPI_GELU | EPI_F16);

  // block 1
  pos_att(m2, pa1_r, wT[2], act2, act1, 0.30f);
  gemm256(act1, wT[7], mlp1_b1, nullptr, nullptr, act0, EPI_BIAS | EPI_GELU | EPI_F16);
  gemm256(act2, wT[9], w1_b,    nullptr, g0,      nullptr, EPI_BIAS | EPI_F32);
  gemm256(act0, wT[8], mlp1_b2, g0,      nullptr, act1, EPI_BIAS | EPI_ADD | EPI_GELU | EPI_F16);

  // up pos_att + decoder
  pos_att(m3, up_r, wT[3], act1, act2, 0.80f);
  gemm256(act2, wT[10], de1_b, nullptr, nullptr, act0, EPI_BIAS | EPI_GELU | EPI_F16);
  k_de2<<<32, 256, 0, stream>>>(act0, de2_w, de2_b, (float*)d_out);
}
